// DGCNN_73950746902596
// MI455X (gfx1250) — matrix operand order repaired and checked by reading, not compiled, not run on hardware
//
#include <hip/hip_runtime.h>
#include <hip/hip_bf16.h>

#define BB   16
#define PP   2048
#define KNN  20
#define EPSBN 1e-5f

typedef __attribute__((ext_vector_type(16))) _Float16 v16h;
typedef __attribute__((ext_vector_type(8)))  _Float16 v8h;
typedef __attribute__((ext_vector_type(8)))  float    v8f;
typedef _Float16 h16;

// ---------------- weight pack: f32 [Kd,N] -> f16 WMMA-B-fragment order -----------------
// Wf[((tile*steps + s)*32 + lane)*16 + i] = W[k,n],  k = s*32 + (lane>=16)*16 + i,
//                                           n = tile*16 + (lane&15); zero-padded.
__global__ void pack_w_kernel(const float* __restrict__ W, h16* __restrict__ Wf,
                              int Kd, int N, int steps, int tot) {
  int t = blockIdx.x * blockDim.x + threadIdx.x;
  if (t >= tot) return;
  int i    = t & 15;
  int L    = (t >> 4) & 31;
  int rest = t >> 9;
  int s    = rest % steps;
  int tile = rest / steps;
  int n = tile * 16 + (L & 15);
  int k = s * 32 + ((L >> 4) << 4) + i;
  Wf[t] = (k < Kd && n < N) ? (h16)W[(size_t)k * N + n] : (h16)0.f;
}

// ---------------- kNN: per-point top-20 (excluding self), insertion sort in regs -------
template <int C>
__global__ __launch_bounds__(128)
void knn_kernel(const float* __restrict__ x, int* __restrict__ idx) {
  constexpr int TILE = 128;
  __shared__ float sx[TILE * C];
  int b = blockIdx.y;
  int i = blockIdx.x * 128 + threadIdx.x;
  const float* xb = x + (size_t)b * PP * C;

  float xi[C];
#pragma unroll
  for (int c = 0; c < C; ++c) xi[c] = xb[(size_t)i * C + c];

  float bd[KNN]; int bi[KNN];
#pragma unroll
  for (int s = 0; s < KNN; ++s) { bd[s] = 3.0e38f; bi[s] = 0; }

  for (int t = 0; t < PP; t += TILE) {
    __syncthreads();
#pragma unroll
    for (int c = 0; c < C; ++c)
      sx[threadIdx.x * C + c] = xb[(size_t)(t + threadIdx.x) * C + c];
    __syncthreads();

    for (int j = 0; j < TILE; ++j) {
      int jj = t + j;
      float d = 0.f;
#pragma unroll
      for (int c = 0; c < C; ++c) {
        float df = xi[c] - sx[j * C + c];
        d = fmaf(df, df, d);
      }
      if (jj == i) d = 1.0e30f;             // mask self (ref adds 1e10 on diagonal)
      if (d < bd[KNN - 1]) {
#pragma unroll
        for (int s = KNN - 1; s > 0; --s) {
          bool shift = d < bd[s - 1];
          float nd = shift ? bd[s - 1] : d;
          int   ni = shift ? bi[s - 1] : jj;
          if (d < bd[s]) { bd[s] = nd; bi[s] = ni; }
        }
        if (d < bd[0]) { bd[0] = d; bi[0] = jj; }
      }
    }
  }
  int* ip = idx + ((size_t)b * PP + i) * KNN;
#pragma unroll
  for (int s = 0; s < KNN; ++s) ip[s] = bi[s];
}

// ---------------- EdgeConv feature build: feat = [x_i | x_j - x_i | 0-pad] -------------
__global__ void build_feat_kernel(const float* __restrict__ x, const int* __restrict__ idx,
                                  h16* __restrict__ feat, int C, int Kpad, int total) {
  int t = blockIdx.x * blockDim.x + threadIdx.x;
  if (t >= total) return;           // total = B*P*K * Kpad
  int cc = t % Kpad;
  int e  = t / Kpad;                // flattened (b,p,k)
  int bp = e / KNN;
  int b  = bp / PP;
  float v;
  if (cc < C) {
    v = x[(size_t)bp * C + cc];
  } else if (cc < 2 * C) {
    int c = cc - C;
    int j = idx[e];
    v = x[((size_t)b * PP + j) * C + c] - x[(size_t)bp * C + c];
  } else {
    v = 0.f;                        // K-dim zero pad (conv1: 6 -> 32)
  }
  feat[t] = (h16)v;
}

// ---------------- WMMA GEMM: C[M,N] = A[M,Kd32] * W[Kd32,Npad] + bias ------------------
// A rows zero-padded to Kd32 (mult of 32); W pre-packed in fragment order.
// Wave computes 64x16 of C (4 M-tiles): one B fragment feeds 4 WMMAs per K-step.
// K-loop is register double-buffered (ping-pong) so load waits are partial.
__global__ __launch_bounds__(128)
void gemm_wmma_kernel(const h16* __restrict__ A, const h16* __restrict__ Wf,
                      const float* __restrict__ bias, float* __restrict__ Cout,
                      int M, int N, int Kd32) {
  int lane = threadIdx.x & 31;
  int wave = threadIdx.x >> 5;
  int l16  = lane & 15;
  int half = lane >> 4;
  int tm = (blockIdx.y * 4 + wave) * 64;
  int tn = blockIdx.x;                         // N-tile index
  if (tm >= M) return;                         // uniform per wave

  int steps = Kd32 >> 5;
  const h16* ap0 = A + (size_t)(tm + l16) * Kd32 + (half << 4);
  const h16* bp  = Wf + ((size_t)tn * steps * 32 + lane) * 16;
  const size_t rstride = (size_t)16 * Kd32;    // 16 rows per M-tile

  v8f acc[4];
#pragma unroll
  for (int m = 0; m < 4; ++m) acc[m] = (v8f){0.f, 0.f, 0.f, 0.f, 0.f, 0.f, 0.f, 0.f};

  v8h alo0[4], ahi0[4], alo1[4], ahi1[4];
  v16h b0, b1;

  auto load0 = [&](int s) {
    b0 = *(const v16h*)(bp + (size_t)s * 512);
#pragma unroll
    for (int m = 0; m < 4; ++m) {
      const h16* p = ap0 + (size_t)m * rstride + (size_t)s * 32;
      alo0[m] = *(const v8h*)(p);
      ahi0[m] = *(const v8h*)(p + 8);
    }
  };
  auto load1 = [&](int s) {
    b1 = *(const v16h*)(bp + (size_t)s * 512);
#pragma unroll
    for (int m = 0; m < 4; ++m) {
      const h16* p = ap0 + (size_t)m * rstride + (size_t)s * 32;
      alo1[m] = *(const v8h*)(p);
      ahi1[m] = *(const v8h*)(p + 8);
    }
  };
  auto mm0 = [&]() {
#pragma unroll
    for (int m = 0; m < 4; ++m) {
      v16h a = __builtin_shufflevector(alo0[m], ahi0[m], 0, 1, 2, 3, 4, 5, 6, 7,
                                       8, 9, 10, 11, 12, 13, 14, 15);
      acc[m] = __builtin_amdgcn_wmma_f32_16x16x32_f16(false, a, false, b0,
                                                      (short)0, acc[m], false, false);
    }
  };
  auto mm1 = [&]() {
#pragma unroll
    for (int m = 0; m < 4; ++m) {
      v16h a = __builtin_shufflevector(alo1[m], ahi1[m], 0, 1, 2, 3, 4, 5, 6, 7,
                                       8, 9, 10, 11, 12, 13, 14, 15);
      acc[m] = __builtin_amdgcn_wmma_f32_16x16x32_f16(false, a, false, b1,
                                                      (short)0, acc[m], false, false);
    }
  };

  load0(0);
  if (steps > 1) {                             // steps is even at all multi-step sites
    int s = 0;
    for (; s + 2 < steps; s += 2) {
      load1(s + 1);                            // prefetch while consuming buf0
      mm0();
      load0(s + 2);                            // prefetch while consuming buf1
      mm1();
    }
    load1(steps - 1);
    mm0();
    mm1();
  } else {
    mm0();
  }

  int col = tn * 16 + l16;
  if (col < N) {
    float bv = bias ? bias[col] : 0.f;
#pragma unroll
    for (int m = 0; m < 4; ++m) {
#pragma unroll
      for (int r = 0; r < 8; ++r) {
        int row = tm + m * 16 + (half << 3) + r;
        Cout[(size_t)row * N + col] = acc[m][r] + bv;
      }
    }
  }
}

// ---------------- BatchNorm helpers ----------------------------------------------------
__global__ void zero_kernel(float* p, int n) {
  int i = blockIdx.x * blockDim.x + threadIdx.x;
  if (i < n) p[i] = 0.f;
}

__global__ __launch_bounds__(256)
void bn_stats_kernel(const float* __restrict__ h, float* __restrict__ stats,
                     int M, int N, int rchunk, int leaky) {
  int c = blockIdx.x * 256 + threadIdx.x;
  if (c >= N) return;
  int r0 = blockIdx.y * rchunk;
  int r1 = r0 + rchunk; if (r1 > M) r1 = M;
  float s = 0.f, s2 = 0.f;
  for (int r = r0; r < r1; ++r) {
    float v = h[(size_t)r * N + c];
    if (leaky) v = (v > 0.f) ? v : 0.01f * v;
    s += v;
    s2 = fmaf(v, v, s2);
  }
  atomicAdd(&stats[c], s);
  atomicAdd(&stats[N + c], s2);
}

__global__ void bn_apply_kernel(const float* __restrict__ h, const float* __restrict__ stats,
                                const float* __restrict__ g, const float* __restrict__ bt,
                                h16* __restrict__ out, int M, int N, int leaky, int postrelu) {
  int t = blockIdx.x * blockDim.x + threadIdx.x;
  if (t >= M * N) return;
  int c = t % N;
  float inv  = 1.f / (float)M;
  float mean = stats[c] * inv;
  float var  = stats[N + c] * inv - mean * mean;   // biased var (ddof=0), matches jnp.var
  float sc   = g[c] * rsqrtf(var + EPSBN);
  float v = h[t];
  if (leaky) v = (v > 0.f) ? v : 0.01f * v;
  v = (v - mean) * sc + bt[c];
  if (postrelu) v = fmaxf(v, 0.f);
  out[t] = (h16)v;
}

__global__ void bn_max_kernel(const float* __restrict__ h, const float* __restrict__ stats,
                              const float* __restrict__ g, const float* __restrict__ bt,
                              float* __restrict__ xout, int N) {
  int t = blockIdx.x * blockDim.x + threadIdx.x;   // over B*P*N
  if (t >= BB * PP * N) return;
  int c  = t % N;
  int bp = t / N;
  float inv  = 1.f / (float)(BB * PP * KNN);
  float mean = stats[c] * inv;
  float var  = stats[N + c] * inv - mean * mean;
  float sc   = g[c] * rsqrtf(var + EPSBN);
  float btc  = bt[c];
  float m = -3.0e38f;
  for (int k = 0; k < KNN; ++k) {
    float v = h[((size_t)bp * KNN + k) * N + c];
    m = fmaxf(m, (v - mean) * sc + btc);
  }
  xout[t] = m;
}

__global__ void concat_kernel(const float* __restrict__ x1, const float* __restrict__ x2,
                              const float* __restrict__ x3, const float* __restrict__ x4,
                              h16* __restrict__ out) {
  int t = blockIdx.x * blockDim.x + threadIdx.x;
  if (t >= BB * PP * 320) return;
  int c = t % 320, r = t / 320;
  float v;
  if      (c <  64) v = x1[r *  64 + c];
  else if (c < 128) v = x2[r *  64 + (c -  64)];
  else if (c < 192) v = x3[r *  64 + (c - 128)];
  else              v = x4[r * 128 + (c - 192)];
  out[t] = (h16)v;
}

static inline int ceildiv(int a, int b) { return (a + b - 1) / b; }

extern "C" void kernel_launch(void* const* d_in, const int* in_sizes, int n_in,
                              void* d_out, int out_size, void* d_ws, size_t ws_size,
                              hipStream_t stream) {
  (void)in_sizes; (void)n_in; (void)out_size; (void)ws_size;
  const float* P[48];
  for (int i = 0; i < 48; ++i) P[i] = (const float*)d_in[i];
  const float* x0 = P[0];                    // [B*P, 3]

  // -------- workspace layout (peak ~890 MB) --------
  char* ws = (char*)d_ws;
  const size_t MB = 1ull << 20;
  h16*   w16   = (h16*)(ws);                 //   0.. 4 MB : packed f16 weights
  int*   idx   = (int*)(ws + 4 * MB);        //   4.. 8 MB : kNN indices
  float* stats = (float*)(ws + 8 * MB);      //   8.. 9 MB : BN sums/sumsq
  float* x1    = (float*)(ws + 9 * MB);
  float* x2    = (float*)(ws + 17 * MB);
  float* x3    = (float*)(ws + 25 * MB);
  float* x4    = (float*)(ws + 33 * MB);
  h16*   feat  = (h16*)(ws + 49 * MB);       //  49..385 MB : f16 GEMM-A (max 655360x128)
  float* hbuf  = (float*)(ws + 385 * MB);    // 385..721 MB : f32 pre-BN activations
  h16*   act   = (h16*)(ws + 721 * MB);      // 721..889 MB : f16 normalized activations

  // -------- pack all weights into WMMA-fragment order (zero-padded) --------
  size_t woff = 0;
  auto packw = [&](const float* w, int Kd, int N) -> h16* {
    int K32   = ((Kd + 31) / 32) * 32;
    int steps = K32 / 32;
    int NT    = (N + 15) / 16;
    int tot   = NT * steps * 512;            // 32 lanes * 16 halves per (tile,step)
    h16* dst = w16 + woff;
    pack_w_kernel<<<ceildiv(tot, 256), 256, 0, stream>>>(w, dst, Kd, N, steps, tot);
    woff += (size_t)tot;
    return dst;
  };
  h16* W11 = packw(P[2],    6,   64);  h16* W12 = packw(P[6],   64,  64);
  h16* W21 = packw(P[10], 128,   64);  h16* W22 = packw(P[14],  64,  64);
  h16* W31 = packw(P[18], 128,   64);  h16* W32 = packw(P[22],  64,  64);
  h16* W41 = packw(P[26], 128,  128);  h16* W42 = packw(P[30], 128, 128);
  h16* W5  = packw(P[34], 320, 1024);
  h16* W6  = packw(P[38], 1024, 512);
  h16* W7  = packw(P[42], 512,  256);
  h16* W8  = packw(P[46], 256,   40);

  // -------- EdgeConv layer --------
  auto edgeconv = [&](const float* xin, int Cin, int Cout,
                      const h16* W1h, const float* b1, const float* g1, const float* bt1,
                      const h16* W2h, const float* b2, const float* g2, const float* bt2,
                      float* xout) {
    dim3 kg(PP / 128, BB);
    if (Cin == 3) knn_kernel<3><<<kg, 128, 0, stream>>>(xin, idx);
    else          knn_kernel<64><<<kg, 128, 0, stream>>>(xin, idx);

    int M = BB * PP * KNN;                                  // 655360 (mult of 256)
    int Kpad = ((2 * Cin + 31) / 32) * 32;                  // conv1: 32; else 2*Cin
    int total = M * Kpad;
    build_feat_kernel<<<ceildiv(total, 256), 256, 0, stream>>>(xin, idx, feat, Cin,
                                                               Kpad, total);

    dim3 gg(ceildiv(Cout, 16), M / 256);
    // MLP1: [M,Kpad] x [Kpad,Cout]
    gemm_wmma_kernel<<<gg, 128, 0, stream>>>(feat, W1h, b1, hbuf, M, Cout, Kpad);
    // BN1 (stats on raw h) -> relu(bn(h)) -> f16
    zero_kernel<<<ceildiv(2 * Cout, 256), 256, 0, stream>>>(stats, 2 * Cout);
    dim3 sg(ceildiv(Cout, 256), ceildiv(M, 2048));
    bn_stats_kernel<<<sg, 256, 0, stream>>>(hbuf, stats, M, Cout, 2048, 0);
    bn_apply_kernel<<<ceildiv(M * Cout, 256), 256, 0, stream>>>(hbuf, stats, g1, bt1,
                                                                act, M, Cout, 0, 1);
    // MLP2: [M,Cout] x [Cout,Cout]
    gemm_wmma_kernel<<<gg, 128, 0, stream>>>(act, W2h, b2, hbuf, M, Cout, Cout);
    // BN2 + max over K
    zero_kernel<<<ceildiv(2 * Cout, 256), 256, 0, stream>>>(stats, 2 * Cout);
    bn_stats_kernel<<<sg, 256, 0, stream>>>(hbuf, stats, M, Cout, 2048, 0);
    bn_max_kernel<<<ceildiv(BB * PP * Cout, 256), 256, 0, stream>>>(hbuf, stats, g2, bt2,
                                                                    xout, Cout);
  };

  edgeconv(x0,  3,  64, W11, P[3],  P[4],  P[5],  W12, P[7],  P[8],  P[9],  x1);
  edgeconv(x1, 64,  64, W21, P[11], P[12], P[13], W22, P[15], P[16], P[17], x2);
  edgeconv(x2, 64,  64, W31, P[19], P[20], P[21], W32, P[23], P[24], P[25], x3);
  edgeconv(x3, 64, 128, W41, P[27], P[28], P[29], W42, P[31], P[32], P[33], x4);

  // -------- head: concat -> conv5 -> fc1 -> fc2 -> fc3 --------
  concat_kernel<<<ceildiv(BB * PP * 320, 256), 256, 0, stream>>>(x1, x2, x3, x4, feat);

  int M = BB * PP;                                          // 32768 (mult of 256)
  auto fc = [&](const h16* Ain, const h16* Wh, const float* b,
                const float* g, const float* bt, int N, int Kd, h16* actout) {
    dim3 gg(ceildiv(N, 16), M / 256);
    gemm_wmma_kernel<<<gg, 128, 0, stream>>>(Ain, Wh, b, hbuf, M, N, Kd);
    zero_kernel<<<ceildiv(2 * N, 256), 256, 0, stream>>>(stats, 2 * N);
    dim3 sg(ceildiv(N, 256), ceildiv(M, 2048));
    bn_stats_kernel<<<sg, 256, 0, stream>>>(hbuf, stats, M, N, 2048, 1);   // bn(leaky(h))
    bn_apply_kernel<<<ceildiv(M * N, 256), 256, 0, stream>>>(hbuf, stats, g, bt,
                                                             actout, M, N, 1, 0);
  };

  fc(feat, W5, P[35], P[36], P[37], 1024, 320, act);        // conv5
  fc(act,  W6, P[39], P[40], P[41],  512, 1024, act);       // fc1
  fc(act,  W7, P[43], P[44], P[45],  256,  512, act);       // fc2

  // fc3: plain linear into d_out (f32, [32768,40])
  dim3 gg(ceildiv(40, 16), M / 256);
  gemm_wmma_kernel<<<gg, 128, 0, stream>>>(act, W8, P[47], (float*)d_out, M, 40, 256);
}
